// MultiheadSelfAttention_14903536517696
// MI455X (gfx1250) — matrix operand order repaired and checked by reading, not compiled, not run on hardware
//
#include <hip/hip_runtime.h>
#include <hip/hip_bf16.h>

typedef __attribute__((ext_vector_type(16))) _Float16 v16h;
typedef __attribute__((ext_vector_type(8)))  _Float16 v8h;
typedef __attribute__((ext_vector_type(8)))  float    v8f;

#define S_LEN   2048
#define D_MODEL 1024
#define NHEAD   16
#define DH      64
#define BATCH   2
#define M_TOT   (BATCH * S_LEN)   // 4096

__device__ __forceinline__ v16h cat8(v8h lo, v8h hi) {
  return __builtin_shufflevector(lo, hi, 0,1,2,3,4,5,6,7,8,9,10,11,12,13,14,15);
}

// A fragment (16x32 f16, M x K), source row-major [m][k], ld in halves.
// Lane 0-15: row m=lane, K {0..7, 16..23}; lane 16-31: row m=lane-16, K {8..15, 24..31}.
__device__ __forceinline__ v16h frag_a(const _Float16* base, int ld, int lane) {
  const int m  = lane & 15;
  const int k0 = (lane & 16) ? 8 : 0;
  const _Float16* p = base + m * ld + k0;
  v8h lo = *(const v8h*)(p);
  v8h hi = *(const v8h*)(p + 16);
  return cat8(lo, hi);
}

// B fragment (32x16 f16, K x N) from transposed-B storage Bt[n][k] (row n contiguous in k).
// Lane 0-15: col n=lane, K {0..7, 16..23}; lane 16-31: col n=lane-16, K {8..15, 24..31}.
__device__ __forceinline__ v16h frag_bt(const _Float16* base, int ld, int lane) {
  const int n  = lane & 15;
  const int k0 = (lane & 16) ? 8 : 0;
  const _Float16* p = base + n * ld + k0;
  v8h lo = *(const v8h*)(p);
  v8h hi = *(const v8h*)(p + 16);
  return cat8(lo, hi);
}

__device__ __forceinline__ v8f wmma16(v16h a, v16h b, v8f c) {
  return __builtin_amdgcn_wmma_f32_16x16x32_f16(false, a, false, b, (short)0, c, false, false);
}

__device__ __forceinline__ v8h cvt8(float4 a, float4 b) {
  v8h h;
  h[0]=(_Float16)a.x; h[1]=(_Float16)a.y; h[2]=(_Float16)a.z; h[3]=(_Float16)a.w;
  h[4]=(_Float16)b.x; h[5]=(_Float16)b.y; h[6]=(_Float16)b.z; h[7]=(_Float16)b.w;
  return h;
}

// ---------------------------------------------------------------------------
// Kernel 1: Y = X * W^T (f32 in, f16 out), stored per-head [b][h][s][dh].
// grid = (M/256, N/64, 3) ; block = 256 (8 waves, 32x64 output per wave)
// which = 0:Q (softmax scale folded in), 1:K, 2:V
// ---------------------------------------------------------------------------
__global__ __launch_bounds__(256) void qkv_proj_kernel(
    const float* __restrict__ x,
    const float* __restrict__ Wq, const float* __restrict__ Wk,
    const float* __restrict__ Wv,
    _Float16* __restrict__ Qb, _Float16* __restrict__ Kb, _Float16* __restrict__ Vb)
{
  __shared__ _Float16 Xs[256 * 40];
  __shared__ _Float16 Ws[64 * 40];

  const int which = blockIdx.z;
  const float* W  = (which == 0) ? Wq : (which == 1) ? Wk : Wv;
  _Float16* Out   = (which == 0) ? Qb : (which == 1) ? Kb : Vb;
  const float scale = (which == 0) ? 0.125f : 1.0f;   // 1/sqrt(64) folded into Q

  const int mbase = blockIdx.x * 256;
  const int nbase = blockIdx.y * 64;
  const int tid  = threadIdx.x;
  const int lane = tid & 31;
  const int wave = tid >> 5;

  v8f acc[2][4] = {};

  const int r8 = tid >> 2;          // 0..63
  const int c8 = (tid & 3) * 8;     // 0,8,16,24

  for (int kc = 0; kc < D_MODEL; kc += 32) {
    // Stage X tile (256x32) f32 -> f16
#pragma unroll
    for (int rr = 0; rr < 256; rr += 64) {
      const float* gp = x + (size_t)(mbase + r8 + rr) * D_MODEL + kc + c8;
      float4 f0 = *(const float4*)(gp);
      float4 f1 = *(const float4*)(gp + 4);
      *(v8h*)&Xs[(r8 + rr) * 40 + c8] = cvt8(f0, f1);
      if (kc + 32 < D_MODEL) __builtin_prefetch(gp + 32, 0, 1);  // global_prefetch_b8
    }
    // Stage W tile (64x32): Bt[n][k] = W[n][k] (W is row-major [out][in])
    {
      const float* gw = W + (size_t)(nbase + r8) * D_MODEL + kc + c8;
      float4 f0 = *(const float4*)(gw);
      float4 f1 = *(const float4*)(gw + 4);
      *(v8h*)&Ws[r8 * 40 + c8] = cvt8(f0, f1);
      if (kc + 32 < D_MODEL) __builtin_prefetch(gw + 32, 0, 1);
    }
    __syncthreads();

    v16h a0 = frag_a(&Xs[(wave * 32) * 40], 40, lane);
    v16h a1 = frag_a(&Xs[(wave * 32 + 16) * 40], 40, lane);
#pragma unroll
    for (int nt = 0; nt < 4; ++nt) {
      v16h b = frag_bt(&Ws[(nt * 16) * 40], 40, lane);
      acc[0][nt] = wmma16(a0, b, acc[0][nt]);
      acc[1][nt] = wmma16(a1, b, acc[1][nt]);
    }
    __syncthreads();
  }

  // Epilogue: store f16 into [b][h][s][dh]
  const int mhi = (lane & 16) ? 8 : 0;
#pragma unroll
  for (int mt = 0; mt < 2; ++mt) {
    const int m0 = mbase + wave * 32 + mt * 16 + mhi;
#pragma unroll
    for (int nt = 0; nt < 4; ++nt) {
      const int n = nbase + nt * 16 + (lane & 15);
      const int h = n >> 6;
      const int d = n & 63;
#pragma unroll
      for (int r = 0; r < 8; ++r) {
        const int m = m0 + r;
        const int b = m >> 11;        // /2048
        const int s = m & 2047;
        const size_t off = ((size_t)((b * NHEAD + h) * S_LEN + s)) * DH + d;
        Out[off] = (_Float16)(acc[mt][nt][r] * scale);
      }
    }
  }
}

// ---------------------------------------------------------------------------
// Kernel 2: causal flash attention. grid = (B*H, S/128), block = 256.
// Each wave: 16 query rows, full dh=64 accumulator. K-blocks of 32 keys.
// Output: attn f16 in [b][s][h*64+d] layout (ready for output projection).
// ---------------------------------------------------------------------------
__global__ __launch_bounds__(256) void flash_attn_kernel(
    const _Float16* __restrict__ Qb, const _Float16* __restrict__ Kb,
    const _Float16* __restrict__ Vb, _Float16* __restrict__ Ab)
{
  __shared__ _Float16 Ks[32 * 72];        // K rows  [key][d]
  __shared__ _Float16 Vt[64 * 40];        // V^T     [d][key]
  __shared__ _Float16 Ps[8 * 16 * 40];    // per-wave P tile [m][k]

  const int bh   = blockIdx.x;            // 0..31
  const int qblk = blockIdx.y;            // 0..15
  const int tid  = threadIdx.x;
  const int lane = tid & 31;
  const int wave = tid >> 5;
  const size_t base = (size_t)bh * S_LEN * DH;

  const int qbase_w = qblk * 128 + wave * 16;

  // Q A-fragments (2 chunks of K=32 over dh=64), kept in registers.
  const int k0a = (lane & 16) ? 8 : 0;
  const _Float16* qrow = Qb + base + (size_t)(qbase_w + (lane & 15)) * DH;
  v16h qa[2];
#pragma unroll
  for (int j = 0; j < 2; ++j) {
    v8h lo = *(const v8h*)(qrow + j * 32 + k0a);
    v8h hi = *(const v8h*)(qrow + j * 32 + k0a + 16);
    qa[j] = cat8(lo, hi);
  }

  v8f o[4] = {};
  float mrun[8], lrun[8];
#pragma unroll
  for (int r = 0; r < 8; ++r) { mrun[r] = -3.0e38f; lrun[r] = 0.0f; }

  const int nkb = (qblk + 1) * 4;         // 32-key blocks needed (causal)
  _Float16* pw = &Ps[wave * 16 * 40];
  const int nl = lane & 15;
  const int kh = (lane & 16) ? 8 : 0;
  const int mhi = (lane & 16) ? 8 : 0;

  const int rS = tid >> 3;                // 0..31
  const int cS = (tid & 7) * 8;           // 0..56

  for (int kb = 0; kb < nkb; ++kb) {
    const int kbase = kb * 32;
    // Stage K rows and V transposed (cooperative, whole block).
    {
      const _Float16* kg = Kb + base + (size_t)(kbase + rS) * DH + cS;
      *(v8h*)&Ks[rS * 72 + cS] = *(const v8h*)kg;
      const _Float16* vg = Vb + base + (size_t)(kbase + rS) * DH + cS;
      v8h vv = *(const v8h*)vg;
#pragma unroll
      for (int i = 0; i < 8; ++i) Vt[(cS + i) * 40 + rS] = vv[i];
      if (kb + 1 < nkb) {                 // prefetch next key block into L2/WGP$
        __builtin_prefetch(kg + 32 * DH, 0, 1);
        __builtin_prefetch(vg + 32 * DH, 0, 1);
      }
    }
    __syncthreads();

    if (kbase <= qbase_w + 15) {          // wave-uniform: EXEC stays all-ones
      // S = Q K^T : two 16x16 column tiles, contract over d (2 x K=32).
      v8f s0 = {}, s1 = {};
#pragma unroll
      for (int j = 0; j < 2; ++j) {
        const _Float16* b0p = &Ks[nl * 72 + j * 32 + kh];
        const _Float16* b1p = &Ks[(16 + nl) * 72 + j * 32 + kh];
        v16h b0 = cat8(*(const v8h*)b0p, *(const v8h*)(b0p + 16));
        v16h b1 = cat8(*(const v8h*)b1p, *(const v8h*)(b1p + 16));
        s0 = wmma16(qa[j], b0, s0);
        s1 = wmma16(qa[j], b1, s1);
      }

      // Causal mask + online softmax (row stats live in the C-frag half-wave).
#pragma unroll
      for (int r = 0; r < 8; ++r) {
        const int q = qbase_w + mhi + r;
        float v0 = (kbase + nl      <= q) ? s0[r] : -3.0e38f;
        float v1 = (kbase + 16 + nl <= q) ? s1[r] : -3.0e38f;
        float mx = fmaxf(v0, v1);
        for (int msk = 8; msk >= 1; msk >>= 1) mx = fmaxf(mx, __shfl_xor(mx, msk, 32));
        const float mn    = fmaxf(mrun[r], mx);
        const float alpha = __expf(mrun[r] - mn);
        const float p0 = __expf(v0 - mn);
        const float p1 = __expf(v1 - mn);
        float rs = p0 + p1;
        for (int msk = 8; msk >= 1; msk >>= 1) rs += __shfl_xor(rs, msk, 32);
        lrun[r] = lrun[r] * alpha + rs;
        mrun[r] = mn;
#pragma unroll
        for (int t = 0; t < 4; ++t) o[t][r] *= alpha;
        const int m = mhi + r;
        pw[m * 40 + nl]      = (_Float16)p0;
        pw[m * 40 + 16 + nl] = (_Float16)p1;
      }

      // Wave-internal LDS ordering before re-reading P in A-fragment layout.
      asm volatile("s_wait_dscnt 0" ::: "memory");

      // O += P * V  (contract over 32 keys; 4 d-tiles of 16).
      const _Float16* pr = &pw[nl * 40 + ((lane & 16) ? 8 : 0)];
      v16h pa = cat8(*(const v8h*)pr, *(const v8h*)(pr + 16));
#pragma unroll
      for (int t = 0; t < 4; ++t) {
        const _Float16* vp = &Vt[(t * 16 + nl) * 40 + kh];
        v16h vb = cat8(*(const v8h*)vp, *(const v8h*)(vp + 16));
        o[t] = wmma16(pa, vb, o[t]);
      }
    }
    __syncthreads();
  }

  // Normalize and store attn f16 into [b][s][h*64+d].
  const int b = bh >> 4;
  const int h = bh & 15;
#pragma unroll
  for (int t = 0; t < 4; ++t) {
    const int d = t * 16 + nl;
#pragma unroll
    for (int r = 0; r < 8; ++r) {
      const int q = qbase_w + mhi + r;
      const float val = o[t][r] / lrun[r];
      Ab[((size_t)(b * S_LEN + q)) * D_MODEL + h * DH + d] = (_Float16)val;
    }
  }
}

// ---------------------------------------------------------------------------
// Kernel 3: out = attn * Wo^T (f16 x f16 -> f32). grid = (M/256, N/64).
// ---------------------------------------------------------------------------
__global__ __launch_bounds__(256) void out_proj_kernel(
    const _Float16* __restrict__ A, const float* __restrict__ Wo,
    float* __restrict__ out)
{
  __shared__ _Float16 Xs[256 * 40];
  __shared__ _Float16 Ws[64 * 40];

  const int mbase = blockIdx.x * 256;
  const int nbase = blockIdx.y * 64;
  const int tid  = threadIdx.x;
  const int lane = tid & 31;
  const int wave = tid >> 5;

  v8f acc[2][4] = {};

  const int r8 = tid >> 2;
  const int c8 = (tid & 3) * 8;

  for (int kc = 0; kc < D_MODEL; kc += 32) {
#pragma unroll
    for (int rr = 0; rr < 256; rr += 64) {
      const _Float16* gp = A + (size_t)(mbase + r8 + rr) * D_MODEL + kc + c8;
      *(v8h*)&Xs[(r8 + rr) * 40 + c8] = *(const v8h*)gp;
      if (kc + 32 < D_MODEL) __builtin_prefetch(gp + 32, 0, 1);
    }
    {
      const float* gw = Wo + (size_t)(nbase + r8) * D_MODEL + kc + c8;
      float4 f0 = *(const float4*)(gw);
      float4 f1 = *(const float4*)(gw + 4);
      *(v8h*)&Ws[r8 * 40 + c8] = cvt8(f0, f1);
      if (kc + 32 < D_MODEL) __builtin_prefetch(gw + 32, 0, 1);
    }
    __syncthreads();

    v16h a0 = frag_a(&Xs[(wave * 32) * 40], 40, lane);
    v16h a1 = frag_a(&Xs[(wave * 32 + 16) * 40], 40, lane);
#pragma unroll
    for (int nt = 0; nt < 4; ++nt) {
      v16h b = frag_bt(&Ws[(nt * 16) * 40], 40, lane);
      acc[0][nt] = wmma16(a0, b, acc[0][nt]);
      acc[1][nt] = wmma16(a1, b, acc[1][nt]);
    }
    __syncthreads();
  }

  const int mhi = (lane & 16) ? 8 : 0;
#pragma unroll
  for (int mt = 0; mt < 2; ++mt) {
    const int m0 = mbase + wave * 32 + mt * 16 + mhi;
#pragma unroll
    for (int nt = 0; nt < 4; ++nt) {
      const int n = nbase + nt * 16 + (lane & 15);
#pragma unroll
      for (int r = 0; r < 8; ++r) {
        out[(size_t)(m0 + r) * D_MODEL + n] = acc[mt][nt][r];
      }
    }
  }
}

// ---------------------------------------------------------------------------
extern "C" void kernel_launch(void* const* d_in, const int* in_sizes, int n_in,
                              void* d_out, int out_size, void* d_ws, size_t ws_size,
                              hipStream_t stream) {
  const float* x  = (const float*)d_in[0];
  const float* Wq = (const float*)d_in[1];
  const float* Wk = (const float*)d_in[2];
  const float* Wv = (const float*)d_in[3];
  const float* Wo = (const float*)d_in[4];
  float* out = (float*)d_out;

  _Float16* ws = (_Float16*)d_ws;
  const size_t NELEM = (size_t)M_TOT * D_MODEL;   // 4M halves per tensor
  _Float16* Qb = ws;
  _Float16* Kb = ws + NELEM;
  _Float16* Vb = ws + 2 * NELEM;
  _Float16* Ab = ws + 3 * NELEM;                  // total 32 MB used

  dim3 blk(256, 1, 1);
  dim3 g1(M_TOT / 256, D_MODEL / 64, 3);
  qkv_proj_kernel<<<g1, blk, 0, stream>>>(x, Wq, Wk, Wv, Qb, Kb, Vb);

  dim3 g2(BATCH * NHEAD, S_LEN / 128, 1);
  flash_attn_kernel<<<g2, blk, 0, stream>>>(Qb, Kb, Vb, Ab);

  dim3 g3(M_TOT / 256, D_MODEL / 64, 1);
  out_proj_kernel<<<g3, blk, 0, stream>>>(Ab, Wo, out);
}
